// QuantLSTMCell_11836929868185
// MI455X (gfx1250) — matrix operand order repaired and checked by reading, not compiled, not run on hardware
//
#include <hip/hip_runtime.h>

// ---------------------------------------------------------------------------
// QuantLSTMCell on gfx1250 (CDNA5, wave32, WMMA).
//
//   gates = [x_q | h_q] (4096 x 2048)  @  [W_ih_q | W_hh_q]^T (2048 x 4096)
//
// Precision scheme (exact by construction):
//   * W_q = n * 2^-16 with |n| <= 2048  -> exactly one f16 per weight.
//   * x_q/h_q = hi(f16) + lo(f16) exactly (residual below hi's ulp is on the
//     2^-16 grid and fits f16, incl. subnormals which WMMA preserves).
//   => A@W^T = A_hi@W^T + A_lo@W^T, f32 accumulate: 2x f16 WMMA volume,
//      ~fp32-exact products.
//
// Fragment loads are straight b128 global loads matching the ISA 7.12.2
// per-lane layouts (A 16x32 f16, B 32x16 f16) -- no LDS transpose.
// Epilogue (bias, fq, sigmoid/tanh, c/h update) is fused per-wave; the
// 64 MB gates intermediate never exists.
// ---------------------------------------------------------------------------

typedef _Float16 v16h __attribute__((ext_vector_type(16)));
typedef _Float16 v8h  __attribute__((ext_vector_type(8)));
typedef float    v8f  __attribute__((ext_vector_type(8)));

#define B_ROWS 4096
#define HID    1024
#define KTOT   2048   // I + H
#define A_ELEMS ((size_t)B_ROWS * KTOT)   // 8388608

__device__ __forceinline__ float fq(float x, float scale, float inv_scale) {
  // round-half-even (rintf default), clamp to int32 range, rescale
  float r = rintf(x * scale);
  r = fminf(fmaxf(r, -2147483648.0f), 2147483647.0f);
  return r * inv_scale;
}

__device__ __forceinline__ v8f wmma_f16(v16h a, v16h b, v8f c) {
  return __builtin_amdgcn_wmma_f32_16x16x32_f16(
      /*neg_a=*/false, a, /*neg_b=*/false, b,
      /*c_mod=*/(short)0, c, /*reuse_a=*/false, /*reuse_b=*/false);
}

// ---------------------------------------------------------------------------
// Prepass: quantize + split A = [x|h] into f16 hi/lo planes, W into exact f16.
// ---------------------------------------------------------------------------
__global__ void lstm_prep_kernel(const float* __restrict__ x,
                                 const float* __restrict__ h,
                                 const float* __restrict__ Wih,
                                 const float* __restrict__ Whh,
                                 const int*   __restrict__ fbp,
                                 _Float16* __restrict__ Ahi,
                                 _Float16* __restrict__ Alo,
                                 _Float16* __restrict__ Bw) {
  const float scale = ldexpf(1.0f, fbp[0]);
  const float inv_scale = 1.0f / scale;
  size_t idx = (size_t)blockIdx.x * blockDim.x + threadIdx.x;
  if (idx < A_ELEMS) {
    unsigned r = (unsigned)(idx / KTOT);
    unsigned cc = (unsigned)(idx % KTOT);
    float s = (cc < HID) ? x[(size_t)r * HID + cc]
                         : h[(size_t)r * HID + (cc - HID)];
    float v = fq(s, scale, inv_scale);
    _Float16 hi = (_Float16)v;
    _Float16 lo = (_Float16)(v - (float)hi);
    Ahi[idx] = hi;
    Alo[idx] = lo;
  } else {
    size_t j = idx - A_ELEMS;
    unsigned r = (unsigned)(j / KTOT);
    unsigned cc = (unsigned)(j % KTOT);
    float s = (cc < HID) ? Wih[(size_t)r * HID + cc]
                         : Whh[(size_t)r * HID + (cc - HID)];
    Bw[j] = (_Float16)fq(s, scale, inv_scale);   // exact: |n| <= 2048
  }
}

// ---------------------------------------------------------------------------
// Fused GEMM + LSTM epilogue.
// Grid: x = HID/16 n-tiles, y = B_ROWS/256 m-tiles. Block = 256 thr = 8 waves.
// Wave w owns rows [mbase + 32w, mbase + 32w + 32) as two 16-row bands, and
// the block's single 16-column band replicated over all 4 gates.
// ---------------------------------------------------------------------------
__global__ void __launch_bounds__(256, 1)
lstm_gemm_kernel(const _Float16* __restrict__ Ahi,
                 const _Float16* __restrict__ Alo,
                 const _Float16* __restrict__ Bw,
                 const float* __restrict__ c_in,
                 const float* __restrict__ b_ih,
                 const float* __restrict__ b_hh,
                 const int*   __restrict__ fbp,
                 float* __restrict__ out) {
  const float scale = ldexpf(1.0f, fbp[0]);
  const float inv_scale = 1.0f / scale;

  const int lane  = (int)(threadIdx.x & 31u);
  const int wave  = (int)(threadIdx.x >> 5u);
  const int lmod  = lane & 15;     // row-in-band (A) / column (B,C)
  const int lhalf = lane >> 4;     // K-half selector per ISA layouts

  const int ncol  = (int)blockIdx.x * 16 + lmod;       // 0..HID-1 (per gate)
  const int mbase = (int)blockIdx.y * 256 + wave * 32; // wave's first row

  const int arow0 = mbase + lmod;        // band 0 source row for A frags
  const int arow1 = arow0 + 16;          // band 1

  v8f acc[2][4];
#pragma unroll
  for (int mi = 0; mi < 2; ++mi)
#pragma unroll
    for (int g = 0; g < 4; ++g)
      acc[mi][g] = (v8f){};

  const _Float16* a0h_p = Ahi + (size_t)arow0 * KTOT + lhalf * 8;
  const _Float16* a0l_p = Alo + (size_t)arow0 * KTOT + lhalf * 8;
  const _Float16* a1h_p = Ahi + (size_t)arow1 * KTOT + lhalf * 8;
  const _Float16* a1l_p = Alo + (size_t)arow1 * KTOT + lhalf * 8;
  const _Float16* b_p[4];
#pragma unroll
  for (int g = 0; g < 4; ++g)
    b_p[g] = Bw + (size_t)(g * HID + ncol) * KTOT + lhalf * 8;

  for (int k = 0; k < KTOT; k += 32) {
    // A fragments: 16-bit A 16x32 layout -> elems 0-7 = K[kh*8..+7],
    // elems 8-15 = K[16+kh*8..+7]  (two contiguous b128 loads).
    v16h a0h, a0l, a1h, a1l;
    *(v8h*)&a0h       = *(const v8h*)(a0h_p + k);
    *(((v8h*)&a0h)+1) = *(const v8h*)(a0h_p + k + 16);
    *(v8h*)&a0l       = *(const v8h*)(a0l_p + k);
    *(((v8h*)&a0l)+1) = *(const v8h*)(a0l_p + k + 16);
    *(v8h*)&a1h       = *(const v8h*)(a1h_p + k);
    *(((v8h*)&a1h)+1) = *(const v8h*)(a1h_p + k + 16);
    *(v8h*)&a1l       = *(const v8h*)(a1l_p + k);
    *(((v8h*)&a1l)+1) = *(const v8h*)(a1l_p + k + 16);

    // B fragments: 32x16 layout -> elems 0-7 = K[kh*8..+7], elems 8-15 =
    // K[16+kh*8..+7] of one W row (row-major W == B^T), two b128 loads.
    v16h bf[4];
#pragma unroll
    for (int g = 0; g < 4; ++g) {
      *(v8h*)&bf[g]       = *(const v8h*)(b_p[g] + k);
      *(((v8h*)&bf[g])+1) = *(const v8h*)(b_p[g] + k + 16);
    }

#pragma unroll
    for (int g = 0; g < 4; ++g) {
      acc[0][g] = wmma_f16(a0h, bf[g], acc[0][g]);
      acc[0][g] = wmma_f16(a0l, bf[g], acc[0][g]);
      acc[1][g] = wmma_f16(a1h, bf[g], acc[1][g]);
      acc[1][g] = wmma_f16(a1l, bf[g], acc[1][g]);
    }
  }

  // ---- fused LSTM epilogue (wave-local: all 4 gates present) ----
  float bsum[4];
#pragma unroll
  for (int g = 0; g < 4; ++g)
    bsum[g] = fq(b_ih[g * HID + ncol], scale, inv_scale) +
              fq(b_hh[g * HID + ncol], scale, inv_scale);

  const size_t c_plane = (size_t)B_ROWS * HID;
#pragma unroll
  for (int mi = 0; mi < 2; ++mi) {
    const int rowb = mbase + mi * 16 + 8 * lhalf;  // C layout: m = v + 8*half
#pragma unroll
    for (int v = 0; v < 8; ++v) {
      const int row = rowb + v;
      float gi = fq(acc[mi][0][v] + bsum[0], scale, inv_scale);
      float gf = fq(acc[mi][1][v] + bsum[1], scale, inv_scale);
      float gg = fq(acc[mi][2][v] + bsum[2], scale, inv_scale);
      float go = fq(acc[mi][3][v] + bsum[3], scale, inv_scale);

      float ig = fq(1.0f / (1.0f + expf(-gi)), scale, inv_scale);
      float fg = fq(1.0f / (1.0f + expf(-gf)), scale, inv_scale);
      float gt = fq(tanhf(gg), scale, inv_scale);
      float og = fq(1.0f / (1.0f + expf(-go)), scale, inv_scale);

      float cq = fq(c_in[(size_t)row * HID + ncol], scale, inv_scale);
      float cn = fq(fg * cq + ig * gt, scale, inv_scale);
      float hn = fq(og * tanhf(cn), scale, inv_scale);

      out[(size_t)row * HID + ncol] = hn;                 // h_new
      out[c_plane + (size_t)row * HID + ncol] = cn;       // c_new
    }
  }
}

extern "C" void kernel_launch(void* const* d_in, const int* in_sizes, int n_in,
                              void* d_out, int out_size, void* d_ws, size_t ws_size,
                              hipStream_t stream) {
  (void)in_sizes; (void)n_in; (void)out_size; (void)ws_size;
  const float* x_t  = (const float*)d_in[0];
  const float* h    = (const float*)d_in[1];
  const float* c    = (const float*)d_in[2];
  const float* W_ih = (const float*)d_in[3];
  const float* W_hh = (const float*)d_in[4];
  const float* b_ih = (const float*)d_in[5];
  const float* b_hh = (const float*)d_in[6];
  const int*   fb   = (const int*)d_in[7];

  // Workspace layout: A_hi (16 MB) | A_lo (16 MB) | W f16 (16 MB) = 48 MB,
  // fully L2-resident (192 MB).
  char* ws = (char*)d_ws;
  _Float16* Ahi = (_Float16*)(ws);
  _Float16* Alo = (_Float16*)(ws + A_ELEMS * sizeof(_Float16));
  _Float16* Bw  = (_Float16*)(ws + 2 * A_ELEMS * sizeof(_Float16));

  // Prepass: 2 * 4096 * 2048 elements, one per thread.
  const unsigned prep_threads = 256;
  const unsigned prep_blocks  = (unsigned)((2 * A_ELEMS) / prep_threads);
  lstm_prep_kernel<<<prep_blocks, prep_threads, 0, stream>>>(
      x_t, h, W_ih, W_hh, fb, Ahi, Alo, Bw);

  // Fused GEMM + epilogue: grid (HID/16 n-tiles, B_ROWS/256 m-tiles).
  dim3 grid(HID / 16, B_ROWS / 256, 1);
  lstm_gemm_kernel<<<grid, 256, 0, stream>>>(
      Ahi, Alo, Bw, c, b_ih, b_hh, fb, (float*)d_out);
}
